// BSLinear_71159018160311
// MI455X (gfx1250) — matrix operand order repaired and checked by reading, not compiled, not run on hardware
//
#include <hip/hip_runtime.h>

// ---------------------------------------------------------------------------
// BSLinear (low-rank + additional-term linear) for MI455X / gfx1250.
// Factored evaluation (292 GFLOP vs 586 for materialize-W):
//   B1[1088,4096] = concat( diag(w^2*mask) @ Vh , Vh_add )      (bf16)
//   B2[4096,1088] = concat( U , U_add ) along columns           (bf16)
//   T  [16384,1088] = Xbf16 @ B1^T        (WMMA bf16, f32 acc -> bf16)
//   out[16384,4096] = T @ B2^T            (WMMA bf16, f32 acc -> f32)
// T/B1/B2 are L2-resident (192MB L2). LDS tiles are filled by the Tensor
// Data Mover (one tensor_load_to_lds per tile, TENSORcnt-tracked, HW pad to
// 80B row stride); falls back to per-lane async global->LDS copies.
// ---------------------------------------------------------------------------

typedef __attribute__((ext_vector_type(16))) __bf16 v16bf;
typedef __attribute__((ext_vector_type(8)))  float  v8f;
typedef int      vsi4 __attribute__((__vector_size__(16)));
typedef int      vsi8 __attribute__((__vector_size__(32)));
typedef unsigned vui4 __attribute__((__vector_size__(16)));

union Frag { v16bf v; uint4 q[2]; };
union BF4  { __bf16 b[4]; uint2 u2; };

#if __has_builtin(__builtin_amdgcn_tensor_load_to_lds_d2) || \
    __has_builtin(__builtin_amdgcn_tensor_load_to_lds)
#define USE_TDM 1
#else
#define USE_TDM 0
#endif

// ---- TDM: load a (tile_rows x 64B) 2D tile, row stride K, into LDS with
//      16B padding per 64B row (-> 80B LDS row stride). Wave-level DMA. ------
__device__ __forceinline__ void tdm_load_2d(const void* gsrc, unsigned lds_addr,
                                            unsigned stride_dw, unsigned tile_rows) {
#if USE_TDM
  unsigned long long ga = (unsigned long long)(size_t)gsrc;
  // D# group0: [1:0]=count=1, [63:32]=lds_addr, [120:64]=global_addr, type=2
  vui4 g0 = { 1u,
              lds_addr,
              (unsigned)ga,
              (((unsigned)(ga >> 32)) & 0x01FFFFFFu) | (2u << 30) };
  const unsigned tile0 = 16;             // 16 DWORDs = 64B per row
  const unsigned td0 = tile0;            // tensor_dim0  (== tile: no OOB clip)
  const unsigned td1 = tile_rows;        // tensor_dim1
  const unsigned w0 = (2u << 16)         // data_size = 4B
                    | (1u << 20)         // pad_enable
                    | (3u << 22)         // pad_interval: every 16 DWORDs
                    | (3u << 25);        // pad_amount: 4 DWORDs (16B)
  vsi8 g1 = { (int)w0,
              (int)((td0 & 0xFFFFu) << 16),                          // bits 63:48 td0.lo
              (int)(((td0 >> 16) & 0xFFFFu) | ((td1 & 0xFFFFu) << 16)),
              (int)(((td1 >> 16) & 0xFFFFu) | (tile0 << 16)),        // tile_dim0
              (int)(tile_rows & 0xFFFFu),                            // tile_dim1, tile_dim2=0
              (int)stride_dw,                                        // tensor_dim0_stride lo
              0, 0 };
#if __has_builtin(__builtin_amdgcn_tensor_load_to_lds_d2)
  __builtin_amdgcn_tensor_load_to_lds_d2(g0, g1, 0);
#elif __clang_major__ >= 23
  vsi4 z4 = {0, 0, 0, 0};
  vsi8 z8 = {0, 0, 0, 0, 0, 0, 0, 0};
  __builtin_amdgcn_tensor_load_to_lds(g0, g1, z4, z4, z8, 0);
#else
  vsi4 z4 = {0, 0, 0, 0};
  __builtin_amdgcn_tensor_load_to_lds(g0, g1, z4, z4, 0);
#endif
#endif
}

template<int N>
__device__ __forceinline__ void wait_tensor() {
#if __has_builtin(__builtin_amdgcn_s_wait_tensorcnt)
  __builtin_amdgcn_s_wait_tensorcnt(N);
#else
  asm volatile("s_wait_tensorcnt %0" :: "i"(N) : "memory");
#endif
}

// ---- Fallback: per-lane CDNA5 async global->LDS copy (ASYNCcnt) ------------
__device__ __forceinline__ void async_copy_b128(const __bf16* gsrc, __bf16* lds_dst) {
#if __has_builtin(__builtin_amdgcn_global_load_async_to_lds_b128)
  __builtin_amdgcn_global_load_async_to_lds_b128(
      (__attribute__((address_space(1))) vsi4*)(size_t)gsrc,
      (__attribute__((address_space(3))) vsi4*)(unsigned)(size_t)lds_dst,
      0, 0);
#else
  asm volatile("global_load_async_to_lds_b128 %0, %1, off"
               :: "v"((unsigned)(size_t)lds_dst),
                  "v"((unsigned long long)(size_t)gsrc)
               : "memory");
#endif
}

template<int N>
__device__ __forceinline__ void wait_async() {
#if __has_builtin(__builtin_amdgcn_s_wait_asynccnt)
  __builtin_amdgcn_s_wait_asynccnt(N);
#else
  asm volatile("s_wait_asynccnt %0" :: "i"(N) : "memory");
#endif
}

// ---- X fp32 -> bf16 (one pass; 268MB read + 134MB write) -------------------
__global__ __launch_bounds__(256)
void convert_f32_to_bf16(const float* __restrict__ x,
                         unsigned short* __restrict__ y, int n4) {
  int i = blockIdx.x * 256 + threadIdx.x;
  if (i >= n4) return;
  float4 f = reinterpret_cast<const float4*>(x)[i];
  BF4 o;
  o.b[0] = (__bf16)f.x; o.b[1] = (__bf16)f.y;
  o.b[2] = (__bf16)f.z; o.b[3] = (__bf16)f.w;
  reinterpret_cast<uint2*>(y)[i] = o.u2;
}

// ---- B1[j,i]: j<1024 -> (w[j]^2*mask[j])*Vh[j,i] else Vh_add[j-1024,i] -----
__global__ __launch_bounds__(256)
void prep_b1(const float* __restrict__ w, const float* __restrict__ mask,
             const float* __restrict__ Vh, const float* __restrict__ Vh_add,
             unsigned short* __restrict__ B1) {
  int i = blockIdx.x * 256 + threadIdx.x;      // over 1088*1024 float4 groups
  int row = i >> 10;                           // 4096/4 = 1024 groups per row
  int c4  = i & 1023;
  float s;
  const float* src;
  if (row < 1024) {
    float ww = w[row];
    s = ww * ww * mask[row];
    src = Vh + (size_t)row * 4096;
  } else {
    s = 1.0f;
    src = Vh_add + (size_t)(row - 1024) * 4096;
  }
  float4 f = reinterpret_cast<const float4*>(src)[c4];
  BF4 o;
  o.b[0] = (__bf16)(f.x * s); o.b[1] = (__bf16)(f.y * s);
  o.b[2] = (__bf16)(f.z * s); o.b[3] = (__bf16)(f.w * s);
  reinterpret_cast<uint2*>(B1)[i] = o.u2;
}

// ---- B2[o,j]: j<1024 -> U[o,j] else U_add[o,j-1024] ------------------------
__global__ __launch_bounds__(256)
void prep_b2(const float* __restrict__ U, const float* __restrict__ U_add,
             unsigned short* __restrict__ B2) {
  int i = blockIdx.x * 256 + threadIdx.x;      // over 4096*272 float4 groups
  int row = i / 272;                           // 1088/4 = 272 groups per row
  int c4  = i % 272;
  int j   = c4 * 4;
  float4 f;
  if (j < 1024)
    f = reinterpret_cast<const float4*>(U + (size_t)row * 1024)[c4];
  else
    f = reinterpret_cast<const float4*>(U_add + (size_t)row * 64)[(j - 1024) >> 2];
  BF4 o;
  o.b[0] = (__bf16)f.x; o.b[1] = (__bf16)f.y;
  o.b[2] = (__bf16)f.z; o.b[3] = (__bf16)f.w;
  reinterpret_cast<uint2*>(B2)[i] = o.u2;
}

// ---------------------------------------------------------------------------
// C[M,N] = A[M,K](bf16) @ B[N,K](bf16)^T  via V_WMMA_F32_16X16X32_BF16.
// 256 threads = 8 wave32 in a 2x4 grid; block tile BM x BN, K-step 32.
// LDS double buffered; tiles filled by TDM (wave 0 issues one descriptor per
// tile, others just barrier) or per-lane async copies as fallback. 80B LDS
// row stride -> conflict-free ds_load_b128 fragment gathers.
// ---------------------------------------------------------------------------
template<int BM, int BN, bool OUTF32>
__global__ __launch_bounds__(256)
void gemm_bf16_tn(const unsigned short* __restrict__ Aus,
                  const unsigned short* __restrict__ Bus,
                  float* __restrict__ Cf, unsigned short* __restrict__ Cbus,
                  int M, int N, int K) {
  constexpr int WAVES_M = 2, WAVES_N = 4;
  constexpr int MFRAG = BM / WAVES_M / 16;          // 16x16 tiles per wave (M)
  constexpr int NFRAG = BN / WAVES_N / 16;          // 16x16 tiles per wave (N)
  constexpr int LSTR  = 40;                         // LDS row stride (elems)
  constexpr int ACH   = (BM * 32) / (8 * 256);      // fallback b128 ops/thread
  constexpr int BCH   = (BN * 32) / (8 * 256);
  static_assert(ACH >= 1 && BCH >= 1 && MFRAG >= 1 && NFRAG >= 1, "tiling");

  __shared__ __align__(16) __bf16 As[2][BM * LSTR];
  __shared__ __align__(16) __bf16 Bs[2][BN * LSTR];

  const __bf16* A = reinterpret_cast<const __bf16*>(Aus);
  const __bf16* B = reinterpret_cast<const __bf16*>(Bus);
  __bf16* Cb = reinterpret_cast<__bf16*>(Cbus);

  const int tid  = threadIdx.x;
  const int lane = tid & 31;
  const int wave = tid >> 5;
  const int wm0  = (wave / WAVES_N) * (MFRAG * 16);
  const int wn0  = (wave % WAVES_N) * (NFRAG * 16);
  const int bm0  = blockIdx.y * BM;
  const int bn0  = blockIdx.x * BN;

  const int lrow  = lane & 15;
  const int ahalf = (lane >> 4) * 8;   // A frag: K base 0/8, +16 for hi half
  const int bbase = (lane >> 4) * 8;   // B frag: K base 0/8, +16 for hi half

  v8f zf = {0.f, 0.f, 0.f, 0.f, 0.f, 0.f, 0.f, 0.f};
  v8f acc[MFRAG][NFRAG];
  #pragma unroll
  for (int m = 0; m < MFRAG; ++m)
    #pragma unroll
    for (int n = 0; n < NFRAG; ++n) acc[m][n] = zf;

#if USE_TDM
  // One TDM descriptor per tile (A and B), issued by wave 0 only.
  auto stage = [&](int buf, int k0) {
    if (wave == 0) {
      tdm_load_2d(A + (size_t)bm0 * K + k0, (unsigned)(size_t)&As[buf][0],
                  (unsigned)(K >> 1), BM);
      tdm_load_2d(B + (size_t)bn0 * K + k0, (unsigned)(size_t)&Bs[buf][0],
                  (unsigned)(K >> 1), BN);
    }
  };
#else
  auto stage = [&](int buf, int k0) {
    #pragma unroll
    for (int i = 0; i < ACH; ++i) {
      int c = tid + i * 256;
      int r = c >> 2, col = (c & 3) << 3;
      async_copy_b128(A + (size_t)(bm0 + r) * K + k0 + col,
                      &As[buf][r * LSTR + col]);
    }
    #pragma unroll
    for (int i = 0; i < BCH; ++i) {
      int c = tid + i * 256;
      int r = c >> 2, col = (c & 3) << 3;
      async_copy_b128(B + (size_t)(bn0 + r) * K + k0 + col,
                      &Bs[buf][r * LSTR + col]);
    }
  };
#endif

  stage(0, 0);
  int cur = 0;
  for (int k0 = 0; k0 < K; k0 += 32) {
    const bool hasNext = (k0 + 32) < K;
    if (hasNext) {
      stage(cur ^ 1, k0 + 32);       // prefetch next tile into other buffer
#if USE_TDM
      if (wave == 0) wait_tensor<2>();   // current tile's 2 TDM ops landed
#else
      wait_async<ACH + BCH>();
#endif
    } else {
#if USE_TDM
      if (wave == 0) wait_tensor<0>();
#else
      wait_async<0>();
#endif
    }
    __syncthreads();                 // cur tile visible to all waves

    Frag fa[MFRAG], fb[NFRAG];
    #pragma unroll
    for (int m = 0; m < MFRAG; ++m) {
      const __bf16* p = &As[cur][(wm0 + m * 16 + lrow) * LSTR];
      fa[m].q[0] = *reinterpret_cast<const uint4*>(p + ahalf);
      fa[m].q[1] = *reinterpret_cast<const uint4*>(p + 16 + ahalf);
    }
    #pragma unroll
    for (int n = 0; n < NFRAG; ++n) {
      const __bf16* p = &Bs[cur][(wn0 + n * 16 + lrow) * LSTR];
      fb[n].q[0] = *reinterpret_cast<const uint4*>(p + bbase);
      fb[n].q[1] = *reinterpret_cast<const uint4*>(p + 16 + bbase);
    }
    #pragma unroll
    for (int m = 0; m < MFRAG; ++m)
      #pragma unroll
      for (int n = 0; n < NFRAG; ++n)
        acc[m][n] = __builtin_amdgcn_wmma_f32_16x16x32_bf16(
            false, fa[m].v, false, fb[n].v, (short)0, acc[m][n], false, false);

    if (hasNext) __syncthreads();    // reads of cur done before it is refilled
    cur ^= 1;
  }

  // Epilogue: C/D layout — VGPR g holds M = g + 8*(lane>=16), N = lane&15.
  const int erow = (lane >> 4) * 8;
  #pragma unroll
  for (int m = 0; m < MFRAG; ++m) {
    #pragma unroll
    for (int n = 0; n < NFRAG; ++n) {
      #pragma unroll
      for (int g = 0; g < 8; ++g) {
        size_t r = (size_t)(bm0 + wm0 + m * 16 + erow + g);
        size_t c = (size_t)(bn0 + wn0 + n * 16 + lrow);
        if constexpr (OUTF32) Cf[r * (size_t)N + c] = acc[m][n][g];
        else                  Cb[r * (size_t)N + c] = (__bf16)acc[m][n][g];
      }
    }
  }
}

extern "C" void kernel_launch(void* const* d_in, const int* in_sizes, int n_in,
                              void* d_out, int out_size, void* d_ws, size_t ws_size,
                              hipStream_t stream) {
  const float* X    = (const float*)d_in[0];  // [4,4096,4096]
  const float* w    = (const float*)d_in[1];  // [1024]
  const float* U    = (const float*)d_in[2];  // [4096,1024]
  const float* Vh   = (const float*)d_in[3];  // [1024,4096]
  const float* Uad  = (const float*)d_in[4];  // [4096,64]
  const float* Vhad = (const float*)d_in[5];  // [64,4096]
  const float* mask = (const float*)d_in[6];  // [1024]
  float* out = (float*)d_out;                 // [4,4096,4096] f32

  constexpr int M  = 4 * 4096;   // 16384
  constexpr int K1 = 4096;
  constexpr int N1 = 1024 + 64;  // 1088
  constexpr int N2 = 4096;

  char* ws = (char*)d_ws;
  unsigned short* Xb = (unsigned short*)ws;                                    // M*K1 bf16
  unsigned short* B1 = (unsigned short*)(ws + (size_t)M * K1 * 2);             // N1*K1 bf16
  unsigned short* B2 = (unsigned short*)(ws + (size_t)M * K1 * 2 + (size_t)N1 * K1 * 2);
  unsigned short* T  = (unsigned short*)(ws + (size_t)M * K1 * 2 + 2 * (size_t)N1 * K1 * 2);

  convert_f32_to_bf16<<<(M * K1 / 4 + 255) / 256, 256, 0, stream>>>(X, Xb, M * K1 / 4);
  prep_b1<<<(N1 * K1 / 4) / 256, 256, 0, stream>>>(w, mask, Vh, Vhad, B1);
  prep_b2<<<(N2 * N1 / 4) / 256, 256, 0, stream>>>(U, Uad, B2);

  // T = Xb @ B1^T : M=16384, N=1088, K=4096  (bf16 out, stays in L2)
  gemm_bf16_tn<128, 64, false>
      <<<dim3(N1 / 64, M / 128), 256, 0, stream>>>(Xb, B1, nullptr, T, M, N1, K1);
  // out = T @ B2^T : M=16384, N=4096, K=1088 (f32 out)
  gemm_bf16_tn<128, 128, true>
      <<<dim3(N2 / 128, M / 128), 256, 0, stream>>>(T, B2, out, nullptr, M, N2, N1);
}
